// Compositional_62843961475630
// MI455X (gfx1250) — matrix operand order repaired and checked by reading, not compiled, not run on hardware
//
#include <hip/hip_runtime.h>
#include <hip/hip_bf16.h>

// Problem sizes (fixed by the reference)
#define N_ROWS 4096
#define E_DIM  16384
#define R_DIM  512
#define D_DIM  256

typedef __bf16          v16bf  __attribute__((ext_vector_type(16)));
typedef float           v8f    __attribute__((ext_vector_type(8)));
typedef unsigned short  v16u16 __attribute__((ext_vector_type(16)));

// fp32 -> bf16 with round-to-nearest-even (pure bit math, no libcalls)
__device__ __forceinline__ unsigned short f2bf(float f) {
    unsigned int u = __float_as_uint(f);
    u += 0x7FFFu + ((u >> 16) & 1u);
    return (unsigned short)(u >> 16);
}

// Byte offset inside a 16x32 bf16 A-tile stored in the WMMA A-operand layout.
// Contiguous K order (matches B loads): lanes 0-15 hold M=0-15 K=0..15,
// lanes 16-31 hold M=0-15 K=16..31; each lane owns 32 contiguous bytes.
__device__ __forceinline__ int a_tile_off(int m, int k) {
    int lane = m + ((k & 16) ? 16 : 0);
    int off  = (k & 15) << 1;
    return lane * 32 + off;
}

// Stage a 16x64 fp32 block as two pre-swizzled bf16 A-subtiles in LDS.
// Each of the 512 threads handles elements (sm, sk) and (sm, sk+32).
__device__ __forceinline__ void stage_tile64(const float* __restrict__ src,
                                             int row_base, int ld, int k0,
                                             int sm, int sk,
                                             unsigned short* buf /* 16*64 */) {
    float v0 = src[(size_t)(row_base + sm) * ld + k0 + sk];
    float v1 = src[(size_t)(row_base + sm) * ld + k0 + 32 + sk];
    *(unsigned short*)((char*)buf +        a_tile_off(sm, sk)) = f2bf(v0);
    *(unsigned short*)((char*)buf + 1024 + a_tile_off(sm, sk)) = f2bf(v1);
}

// Stage one fp32 element into a pre-swizzled bf16 A-tile in LDS (fallback).
__device__ __forceinline__ void stage_elem(const float* __restrict__ src,
                                           int row_base, int ld, int k0,
                                           int sm, int sk,
                                           unsigned short* buf) {
    float v = src[(size_t)(row_base + sm) * ld + k0 + sk];
    *(unsigned short*)((char*)buf + a_tile_off(sm, sk)) = f2bf(v);
}

// ---------------------------------------------------------------------------
// One-time streaming conversion: fp32 weights -> bf16 (row-major preserved).
// Row-major k-consecutive bf16 directly gives the contiguous-K B-operand order this file uses.
// ---------------------------------------------------------------------------
__global__ __launch_bounds__(256)
void convert_f32_to_bf16(const float* __restrict__ src,
                         unsigned short* __restrict__ dst, int n) {
    int stride = gridDim.x * blockDim.x * 4;
    for (int i = (blockIdx.x * blockDim.x + threadIdx.x) * 4; i < n; i += stride) {
        float4 q = *(const float4*)(src + i);
        ushort4 o;
        o.x = f2bf(q.x); o.y = f2bf(q.y); o.z = f2bf(q.z); o.w = f2bf(q.w);
        *(ushort4*)(dst + i) = o;
    }
}

#define WMMA_BF16(A, B, C) \
    __builtin_amdgcn_wmma_f32_16x16x32_bf16(false, (A), false, (B), (short)0, (C), false, false)

// ---------------------------------------------------------------------------
// Fast path: weights already bf16 (in workspace). B operand loads are plain
// 32-byte global loads, no per-iteration conversion VALU. Double-buffered
// LDS A-tiles, K=64 per barrier step (4 WMMAs between barriers).
// ---------------------------------------------------------------------------
__global__ __launch_bounds__(512)
void fused_score_kernel_bf16w(const float* __restrict__ x,
                              const float* __restrict__ y,
                              const float* __restrict__ r,
                              const unsigned short* __restrict__ Wb_ent, // [D,E] bf16
                              const unsigned short* __restrict__ Wb_rel, // [D,R] bf16
                              float* __restrict__ out)                   // [N,1]
{
    __shared__ __align__(32) unsigned short Ax[2][16 * 64];
    __shared__ __align__(32) unsigned short Ay[2][16 * 64];
    __shared__ float score[16];

    const int tid      = threadIdx.x;
    const int lane     = tid & 31;
    const int wave     = tid >> 5;                 // 0..15 -> D column tile
    const int row_base = blockIdx.x * 16;

    const int d_col  = wave * 16 + (lane & 15);    // W row this lane loads
    const int kk_sub = (lane >> 4) ? 16 : 0;       // upper half-wave: K+16..+31

    const int sm = wave;                           // staging row
    const int sk = lane;                           // staging k (and k+32)

    if (tid < 16) score[tid] = 0.0f;

    v8f c_r = {0.f,0.f,0.f,0.f,0.f,0.f,0.f,0.f};
    v8f c_a = c_r;
    v8f c_b = c_r;

    // ---------- Phase 1: er = r @ Wrel^T  (K = 512, 8 steps of 64) ----------
    stage_tile64(r, row_base, R_DIM, 0, sm, sk, Ax[0]);
    __syncthreads();
    const unsigned short* wrel_row = Wb_rel + (size_t)d_col * R_DIM + kk_sub;
    for (int it = 0; it < R_DIM / 64; ++it) {
        const int cur = it & 1;
        if (it + 1 < R_DIM / 64)                        // uniform branch
            stage_tile64(r, row_base, R_DIM, (it + 1) * 64, sm, sk, Ax[cur ^ 1]);
        const v16bf* pa = (const v16bf*)Ax[cur];
        v16bf a0 = pa[lane];
        v16bf a1 = pa[32 + lane];
        const unsigned short* wp = wrel_row + it * 64;
        v16bf b0 = *(const v16bf*)wp;
        v16bf b1 = *(const v16bf*)(wp + 32);
        c_r = WMMA_BF16(a0, b0, c_r);
        c_r = WMMA_BF16(a1, b1, c_r);
        __syncthreads();
    }

    // ---------- Phase 2: ea, eb vs Went^T (K = 16384, 256 steps of 64) ------
    stage_tile64(x, row_base, E_DIM, 0, sm, sk, Ax[0]);
    stage_tile64(y, row_base, E_DIM, 0, sm, sk, Ay[0]);
    __syncthreads();
    const unsigned short* went_row = Wb_ent + (size_t)d_col * E_DIM + kk_sub;
    for (int it = 0; it < E_DIM / 64; ++it) {
        const int cur = it & 1;
        if (it + 1 < E_DIM / 64) {                      // uniform branch
            stage_tile64(x, row_base, E_DIM, (it + 1) * 64, sm, sk, Ax[cur ^ 1]);
            stage_tile64(y, row_base, E_DIM, (it + 1) * 64, sm, sk, Ay[cur ^ 1]);
        }
        const v16bf* pax = (const v16bf*)Ax[cur];
        const v16bf* pay = (const v16bf*)Ay[cur];
        v16bf ax0 = pax[lane];
        v16bf ax1 = pax[32 + lane];
        v16bf ay0 = pay[lane];
        v16bf ay1 = pay[32 + lane];

        const unsigned short* wp = went_row + it * 64;
        __builtin_prefetch(wp + 128, 0, 1);             // next-next K chunk
        v16bf b0 = *(const v16bf*)wp;                   // 32B bf16, layout-native
        v16bf b1 = *(const v16bf*)(wp + 32);

        c_a = WMMA_BF16(ax0, b0, c_a);
        c_b = WMMA_BF16(ay0, b0, c_b);
        c_a = WMMA_BF16(ax1, b1, c_a);
        c_b = WMMA_BF16(ay1, b1, c_b);
        __syncthreads();
    }

    // ---------- Epilogue: score[m] = sum_d er*ea*eb, sigmoid ----------
    // C/D layout: VGPR i holds row (i + 8*(lane>=16)), column = lane&15.
    const int rsub = (lane >> 4) ? 8 : 0;
#pragma unroll
    for (int i = 0; i < 8; ++i) {
        float t = c_r[i] * c_a[i] * c_b[i];
        atomicAdd(&score[rsub + i], t);                 // ds_add_f32
    }
    __syncthreads();

    if (tid < 16) {
        float s = score[tid];
        out[row_base + tid] = 1.0f / (1.0f + __expf(-s));
    }
}

// ---------------------------------------------------------------------------
// Fallback path (workspace too small): on-the-fly fp32->bf16 B conversion.
// ---------------------------------------------------------------------------
__device__ __forceinline__ v16bf load_B_f32(const float* __restrict__ wrow) {
    const float4* p = (const float4*)wrow;
    float4 q0 = p[0], q1 = p[1], q2 = p[2], q3 = p[3];
    v16u16 t;
    t[0]  = f2bf(q0.x); t[1]  = f2bf(q0.y); t[2]  = f2bf(q0.z); t[3]  = f2bf(q0.w);
    t[4]  = f2bf(q1.x); t[5]  = f2bf(q1.y); t[6]  = f2bf(q1.z); t[7]  = f2bf(q1.w);
    t[8]  = f2bf(q2.x); t[9]  = f2bf(q2.y); t[10] = f2bf(q2.z); t[11] = f2bf(q2.w);
    t[12] = f2bf(q3.x); t[13] = f2bf(q3.y); t[14] = f2bf(q3.z); t[15] = f2bf(q3.w);
    return __builtin_bit_cast(v16bf, t);
}

__global__ __launch_bounds__(512)
void fused_score_kernel_f32w(const float* __restrict__ x,
                             const float* __restrict__ y,
                             const float* __restrict__ r,
                             const float* __restrict__ Went,
                             const float* __restrict__ Wrel,
                             float* __restrict__ out)
{
    __shared__ __align__(32) unsigned short Ax[16 * 32];
    __shared__ __align__(32) unsigned short Ay[16 * 32];
    __shared__ float score[16];

    const int tid      = threadIdx.x;
    const int lane     = tid & 31;
    const int wave     = tid >> 5;
    const int row_base = blockIdx.x * 16;
    const int d_col    = wave * 16 + (lane & 15);
    const int kk_sub   = (lane >> 4) ? 16 : 0;
    const int sm = wave, sk = lane;

    if (tid < 16) score[tid] = 0.0f;
    __syncthreads();

    v8f c_r = {0.f,0.f,0.f,0.f,0.f,0.f,0.f,0.f};
    v8f c_a = c_r, c_b = c_r;

    for (int k0 = 0; k0 < R_DIM; k0 += 32) {
        stage_elem(r, row_base, R_DIM, k0, sm, sk, Ax);
        __syncthreads();
        v16bf a = ((const v16bf*)Ax)[lane];
        v16bf b = load_B_f32(&Wrel[(size_t)d_col * R_DIM + k0 + kk_sub]);
        c_r = WMMA_BF16(a, b, c_r);
        __syncthreads();
    }

    for (int k0 = 0; k0 < E_DIM; k0 += 32) {
        stage_elem(x, row_base, E_DIM, k0, sm, sk, Ax);
        stage_elem(y, row_base, E_DIM, k0, sm, sk, Ay);
        __syncthreads();
        v16bf ax = ((const v16bf*)Ax)[lane];
        v16bf ay = ((const v16bf*)Ay)[lane];
        const float* wp = &Went[(size_t)d_col * E_DIM + k0 + kk_sub];
        __builtin_prefetch(wp + 32, 0, 1);
        v16bf b = load_B_f32(wp);
        c_a = WMMA_BF16(ax, b, c_a);
        c_b = WMMA_BF16(ay, b, c_b);
        __syncthreads();
    }

    const int rsub = (lane >> 4) ? 8 : 0;
#pragma unroll
    for (int i = 0; i < 8; ++i) {
        float t = c_r[i] * c_a[i] * c_b[i];
        atomicAdd(&score[rsub + i], t);
    }
    __syncthreads();

    if (tid < 16) {
        float s = score[tid];
        out[row_base + tid] = 1.0f / (1.0f + __expf(-s));
    }
}

extern "C" void kernel_launch(void* const* d_in, const int* in_sizes, int n_in,
                              void* d_out, int out_size, void* d_ws, size_t ws_size,
                              hipStream_t stream) {
    (void)in_sizes; (void)n_in; (void)out_size;
    const float* x    = (const float*)d_in[0];
    const float* y    = (const float*)d_in[1];
    const float* r    = (const float*)d_in[2];
    const float* Went = (const float*)d_in[3];
    const float* Wrel = (const float*)d_in[4];
    float* out = (float*)d_out;

    const size_t n_ent = (size_t)D_DIM * E_DIM;          // 4,194,304
    const size_t n_rel = (size_t)D_DIM * R_DIM;          // 131,072
    const size_t need  = (n_ent + n_rel) * sizeof(unsigned short);

    dim3 grid(N_ROWS / 16);   // 256 workgroups of 16 rows
    dim3 block(512);          // 16 waves (wave32) -> 16 D-column tiles

    if (ws_size >= need) {
        unsigned short* Wb_ent = (unsigned short*)d_ws;
        unsigned short* Wb_rel = Wb_ent + n_ent;
        convert_f32_to_bf16<<<dim3(2048), dim3(256), 0, stream>>>(
            Went, Wb_ent, (int)n_ent);
        convert_f32_to_bf16<<<dim3(128), dim3(256), 0, stream>>>(
            Wrel, Wb_rel, (int)n_rel);
        fused_score_kernel_bf16w<<<grid, block, 0, stream>>>(
            x, y, r, Wb_ent, Wb_rel, out);
    } else {
        fused_score_kernel_f32w<<<grid, block, 0, stream>>>(
            x, y, r, Went, Wrel, out);
    }
}
